// RoiHead_42640435315366
// MI455X (gfx1250) — matrix operand order repaired and checked by reading, not compiled, not run on hardware
//
#include <hip/hip_runtime.h>

// ---- problem constants (match reference) ----
#define NROIS   512
#define HFEAT   50
#define WFEAT   75
#define FC_IN   25088          // 512*7*7
#define FC_DIM  4096
#define NLOC    84             // 21*4
#define NSCORE  21

typedef __attribute__((ext_vector_type(16))) __bf16   v16bf;
typedef __attribute__((ext_vector_type(8)))  __bf16   v8bf;
typedef __attribute__((ext_vector_type(8)))  float    v8f;
typedef __attribute__((ext_vector_type(4)))  float    v4f;
typedef __attribute__((ext_vector_type(4)))  unsigned v4u;
typedef __attribute__((ext_vector_type(4)))  int      v4i;
typedef __attribute__((ext_vector_type(8)))  int      v8i;

#if __has_builtin(__builtin_amdgcn_tensor_load_to_lds)
#define USE_TDM 1
#elif __has_builtin(__builtin_amdgcn_global_load_async_to_lds_b128)
#define USE_ASYNC 1
#endif

__device__ __forceinline__ unsigned short bf16_bits(float f) {
  unsigned u = __builtin_bit_cast(unsigned, f);
  u += 0x7fffu + ((u >> 16) & 1u);               // round-to-nearest-even
  return (unsigned short)(u >> 16);
}
__device__ __forceinline__ __bf16 f32_to_bf16(float f) {
  unsigned short h = bf16_bits(f);
  return __builtin_bit_cast(__bf16, h);
}

// ---------------------------------------------------------------------------
// Kernel 1: RoI adaptive max pool -> bf16 [512, 25088], k = c*49 + py*7 + px
// ---------------------------------------------------------------------------
__global__ __launch_bounds__(256) void roi_pool_kernel(
    const float* __restrict__ feat,    // [512,50,75]
    const float* __restrict__ rois,    // [512,4] image coords
    __bf16* __restrict__ pool)
{
  const int roi = blockIdx.x;
  const float* r = rois + roi * 4;
  const int x1 = (int)(r[0] * 0.0625f);
  const int y1 = (int)(r[1] * 0.0625f);
  const int x2 = (int)(r[2] * 0.0625f);
  const int y2 = (int)(r[3] * 0.0625f);
  const int h = y2 - y1 + 1;
  const int w = x2 - x1 + 1;

  for (int o = threadIdx.x; o < FC_IN; o += blockDim.x) {
    const int c  = o / 49;
    const int b  = o - c * 49;
    const int py = b / 7;
    const int px = b - py * 7;
    const int rs = y1 + (py * h) / 7;
    const int re = y1 + ((py + 1) * h + 6) / 7;
    const int cs = x1 + (px * w) / 7;
    const int ce = x1 + ((px + 1) * w + 6) / 7;
    const float* fp = feat + (size_t)c * (HFEAT * WFEAT);
    float m = -INFINITY;
    for (int rr = rs; rr < re; ++rr) {
      const float* rowp = fp + rr * WFEAT;
      for (int cc = cs; cc < ce; ++cc) m = fmaxf(m, rowp[cc]);
    }
    pool[(size_t)roi * FC_IN + o] = f32_to_bf16(m);
  }
}

// ---------------------------------------------------------------------------
// Kernel 2: bf16-WMMA GEMM:  out[M,Nld] = act(A_bf16[M,K] * B_f32[K,Nld] + b)
// Block tile 128(M) x 64(N); 512 thr = 16 waves; wave = 32x16 tile (2 WMMA).
// A tile staged via TDM tensor_load_to_lds (or async-to-LDS / sync fallback).
// NGUARD=false: Nld%64==0 fast path (b128 B loads, b32 ds stores, bf16 out).
// NGUARD=true : ragged-N heads (clamp+select, no exec branches, f32 out).
// ---------------------------------------------------------------------------
template <bool NGUARD>
__global__ __launch_bounds__(512) void gemm_bf16_wmma(
    const __bf16* __restrict__ A,      // [M, K] bf16 row-major
    const float*  __restrict__ B,      // [K, Nld] fp32 row-major
    const float*  __restrict__ bias,   // [Nld]
    __bf16* __restrict__ outBf,        // used when !NGUARD
    float*  __restrict__ outF,         // used when NGUARD
    int K, int Nld)
{
  const int tid  = threadIdx.x;
  const int lane = tid & 31;
  const int wv   = tid >> 5;           // 0..15
  const int wn   = wv & 3;             // N sub-tile (x16)
  const int wm   = wv >> 2;            // M group (x32)
  const int m0   = blockIdx.x * 128;
  const int n0   = blockIdx.y * 64;
  const int hh   = lane >> 4;          // lane half (ISA fragment layout)
  const int l16  = lane & 15;

  __shared__ __align__(16) __bf16 As[128 * 32];   // [m][k]      8 KB
  __shared__ __align__(16) __bf16 Bs[64 * 32];    // [n][k] (T)  4 KB

  v8f acc0 = {};
  v8f acc1 = {};

  for (int k0 = 0; k0 < K; k0 += 32) {
    __syncthreads();

    // ---- stage A tile 128x32 bf16 -> As (row-major, matches TDM layout) ----
#if defined(USE_TDM)
    if (wv == 0) {
      // 2-D Tensor DMA descriptor: data_size=2B, tile 32(k) x 128(m),
      // row stride = K elements; lands contiguously as As[m][k].
      const unsigned lds_off =
          (unsigned)(size_t)(__attribute__((address_space(3))) const void*)As;
      const unsigned long long ga =
          (unsigned long long)(const void*)(A + (size_t)m0 * K + k0);
      const v4u g0 = { 1u,                                  // count=1
                       lds_off,                             // lds_addr
                       (unsigned)(ga & 0xffffffffu),        // global_addr lo
                       (unsigned)((ga >> 32) & 0x01ffffffu) // global_addr hi
                         | (2u << 30) };                    // type=2 (image)
      const v8i g1 = { (int)0x00010000u,    // wg_mask=0, data_size=2B
                       (int)(32u << 16),    // tensor_dim0 = 32
                       (int)(128u << 16),   // tensor_dim1 = 128
                       (int)(32u << 16),    // tile_dim0   = 32
                       128,                 // tile_dim1   = 128
                       K,                   // tensor_dim0_stride (lo32)
                       0, 0 };
      const v4i gz  = { 0, 0, 0, 0 };
      const v8i gz8 = { 0, 0, 0, 0, 0, 0, 0, 0 };
      __builtin_amdgcn_tensor_load_to_lds(g0, g1, gz, gz, gz8, 0);
    }
#elif defined(USE_ASYNC)
    {
      const int row = tid >> 2;
      const int c8  = (tid & 3) * 8;
      __builtin_amdgcn_global_load_async_to_lds_b128(
          (__attribute__((address_space(1))) void*)
              (A + (size_t)(m0 + row) * K + (k0 + c8)),
          (__attribute__((address_space(3))) void*)(As + row * 32 + c8),
          0, 0);
    }
#else
    {
      const int row = tid >> 2;
      const int c8  = (tid & 3) * 8;
      *(v8bf*)(As + row * 32 + c8) =
          *(const v8bf*)(A + (size_t)(m0 + row) * K + (k0 + c8));
    }
#endif

    // ---- stage B tile 32x64 fp32 -> bf16, transposed to Bs[n][k] ----
    if (!NGUARD) {
      if (tid < 256) {                       // wave-uniform predicate
        const int kk = (tid >> 4) * 2;       // 0,2,...,30
        const int n4 = (tid & 15) * 4;
        const float* bp = B + (size_t)(k0 + kk) * Nld + (n0 + n4);
        const v4f r0 = *(const v4f*)bp;
        const v4f r1 = *(const v4f*)(bp + Nld);
        #pragma unroll
        for (int j = 0; j < 4; ++j) {
          const unsigned pk = (unsigned)bf16_bits(r0[j]) |
                              ((unsigned)bf16_bits(r1[j]) << 16);
          *(unsigned*)&Bs[(n4 + j) * 32 + kk] = pk;   // k, k+1 packed
        }
      }
    } else {
      #pragma unroll
      for (int i = 0; i < 4; ++i) {
        const int e  = tid + i * 512;
        const int k  = e >> 6;
        const int n  = e & 63;
        const int ng = n0 + n;
        const int nc = ng < Nld ? ng : Nld - 1;        // clamp: no exec branch
        float v = B[(size_t)(k0 + k) * Nld + nc];
        v = (ng < Nld) ? v : 0.0f;                     // v_cndmask
        Bs[n * 32 + k] = f32_to_bf16(v);
      }
    }
    if (lane == 0 && k0 + 32 < K)            // warm next W tile
      __builtin_prefetch(B + (size_t)(k0 + 32 + wv * 2) * Nld + n0, 0, 1);

#if defined(USE_TDM)
    __builtin_amdgcn_s_wait_tensorcnt(0);
#elif defined(USE_ASYNC)
#if __has_builtin(__builtin_amdgcn_s_wait_asynccnt)
    __builtin_amdgcn_s_wait_asynccnt(0);
#else
    asm volatile("s_wait_asynccnt 0x0" ::: "memory");
#endif
#endif
    __syncthreads();

    // ---- B fragment: 32x16, lane=col, K = 8h+s / 16+8h+(s-8) per lane ----
    v16bf bfr;
    {
      const int ncq = wn * 16 + l16;
      const v8bf blo = *(const v8bf*)(Bs + ncq * 32 + hh * 8);
      const v8bf bhi = *(const v8bf*)(Bs + ncq * 32 + 16 + hh * 8);
      #pragma unroll
      for (int j = 0; j < 8; ++j) { bfr[j] = blo[j]; bfr[j + 8] = bhi[j]; }
    }
    // ---- two A fragments (rows wm*32 .. wm*32+31) + two WMMAs ----
    #pragma unroll
    for (int t = 0; t < 2; ++t) {
      const int rrow = (wm * 2 + t) * 16 + l16;
      const v8bf alo = *(const v8bf*)(As + rrow * 32 + hh * 8);
      const v8bf ahi = *(const v8bf*)(As + rrow * 32 + 16 + hh * 8);
      v16bf afr;
      #pragma unroll
      for (int j = 0; j < 8; ++j) { afr[j] = alo[j]; afr[j + 8] = ahi[j]; }
      if (t == 0)
        acc0 = __builtin_amdgcn_wmma_f32_16x16x32_bf16(
                   false, afr, false, bfr, (short)0, acc0, false, false);
      else
        acc1 = __builtin_amdgcn_wmma_f32_16x16x32_bf16(
                   false, afr, false, bfr, (short)0, acc1, false, false);
    }
  }

  // ---- epilogue: bias + activation; C/D layout: VGPR r -> M = r + 8*hh ----
  const int nc = n0 + wn * 16 + l16;
  if (!NGUARD || nc < Nld) {
    const float bv = bias[nc];
    #pragma unroll
    for (int t = 0; t < 2; ++t) {
      const v8f acc = (t == 0) ? acc0 : acc1;
      const int mbase = m0 + (wm * 2 + t) * 16 + hh * 8;
      #pragma unroll
      for (int r = 0; r < 8; ++r) {
        float v = acc[r] + bv;
        const size_t idx = (size_t)(mbase + r) * Nld + nc;
        if (!NGUARD) outBf[idx] = f32_to_bf16(fmaxf(v, 0.0f));   // ReLU
        else         outF[idx]  = v;                             // heads: linear
      }
    }
  }
}

// ---------------------------------------------------------------------------
extern "C" void kernel_launch(void* const* d_in, const int* in_sizes, int n_in,
                              void* d_out, int out_size, void* d_ws, size_t ws_size,
                              hipStream_t stream) {
  (void)in_sizes; (void)n_in; (void)out_size; (void)ws_size;

  const float* x      = (const float*)d_in[0];   // [1,512,50,75]
  const float* rois   = (const float*)d_in[1];   // [512,4]
  const float* W1     = (const float*)d_in[2];   // [25088,4096]
  const float* b1     = (const float*)d_in[3];
  const float* W2     = (const float*)d_in[4];   // [4096,4096]
  const float* b2     = (const float*)d_in[5];
  const float* W_loc  = (const float*)d_in[6];   // [4096,84]
  const float* b_loc  = (const float*)d_in[7];
  const float* W_sc   = (const float*)d_in[8];   // [4096,21]
  const float* b_sc   = (const float*)d_in[9];

  // workspace: pool (24.5 MB) + fc1 (4 MB) + fc2 (4 MB), all bf16
  __bf16* pool = (__bf16*)d_ws;
  __bf16* fc1  = pool + (size_t)NROIS * FC_IN;
  __bf16* fc2  = fc1  + (size_t)NROIS * FC_DIM;

  float* out_loc = (float*)d_out;                  // [512,84]
  float* out_sc  = out_loc + (size_t)NROIS * NLOC; // [512,21]

  roi_pool_kernel<<<NROIS, 256, 0, stream>>>(x, rois, pool);

  gemm_bf16_wmma<false><<<dim3(NROIS / 128, FC_DIM / 64), 512, 0, stream>>>(
      pool, W1, b1, fc1, nullptr, FC_IN, FC_DIM);

  gemm_bf16_wmma<false><<<dim3(NROIS / 128, FC_DIM / 64), 512, 0, stream>>>(
      fc1, W2, b2, fc2, nullptr, FC_DIM, FC_DIM);

  gemm_bf16_wmma<true><<<dim3(NROIS / 128, (NLOC + 63) / 64), 512, 0, stream>>>(
      fc2, W_loc, b_loc, nullptr, out_loc, FC_DIM, NLOC);

  gemm_bf16_wmma<true><<<dim3(NROIS / 128, (NSCORE + 63) / 64), 512, 0, stream>>>(
      fc2, W_sc, b_sc, nullptr, out_sc, FC_DIM, NSCORE);
}
